// MPNEncoder_7043746365472
// MI455X (gfx1250) — matrix operand order repaired and checked by reading, not compiled, not run on hardware
//
#include <hip/hip_runtime.h>
#include <hip/hip_bf16.h>

// ---------------------------------------------------------------------------
// CDNA5 / gfx1250 WMMA types
// ---------------------------------------------------------------------------
typedef __attribute__((ext_vector_type(16))) __bf16 v16bf;
typedef __attribute__((ext_vector_type(8)))  float  v8f;
typedef __attribute__((ext_vector_type(4)))  float  f32x4;

#define TILE_M 64
#define TILE_N 32
#define TILE_K 32
#define APAD   36   // padded LDS row length in floats (144B: 16B-aligned, conflict-free)

__device__ __forceinline__ float sigmoidf_(float x) {
    return 1.0f / (1.0f + __expf(-x));
}

// ---------------------------------------------------------------------------
// Tile staging into LDS (f32; B pre-transposed to [n][k]).
// Bulk tiles: global_load_async_to_lds_b128 (ASYNCcnt DMA path, no VGPR trip).
// Edge tiles: unconditional loads with clamped addresses + select (no exec
// masking, no branches).
// ---------------------------------------------------------------------------
template<bool BT>
__device__ __forceinline__ void stage_tiles(
    float (&As)[TILE_M][APAD], float (&Bs)[TILE_N][APAD],
    const float* __restrict__ A, int lda,
    const float* __restrict__ B, int ldb,
    int m0, int n0, int k0, int M, int N, int K, int tid)
{
    const bool fullA = (m0 + TILE_M <= M) && (k0 + TILE_K <= K) && ((lda & 3) == 0);
    if (fullA) {
        #pragma unroll
        for (int c = tid; c < 512; c += 256) {          // 512 16B chunks, 2/thread
            int r = c >> 3, q = (c & 7) << 2;
            const float* gp = A + (size_t)(m0 + r) * lda + (k0 + q);
            unsigned la = (unsigned)(size_t)&As[r][q];
            asm volatile("global_load_async_to_lds_b128 %0, %1, off"
                         :: "v"(la), "v"(gp) : "memory");
        }
    } else {
        #pragma unroll
        for (int i = tid; i < TILE_M * TILE_K; i += 256) {
            int r = i >> 5, c = i & 31;
            int gm = m0 + r, gk = k0 + c;
            bool ok = (gm < M) && (gk < K);
            float v = A[(size_t)(ok ? gm : 0) * lda + (ok ? gk : 0)];
            As[r][c] = ok ? v : 0.0f;
        }
    }
    const bool fullB = BT && (n0 + TILE_N <= N) && (k0 + TILE_K <= K) && ((ldb & 3) == 0);
    if (fullB) {
        int r = tid >> 3, q = (tid & 7) << 2;           // 256 16B chunks, 1/thread
        const float* gp = B + (size_t)(n0 + r) * ldb + (k0 + q);
        unsigned la = (unsigned)(size_t)&Bs[r][q];
        asm volatile("global_load_async_to_lds_b128 %0, %1, off"
                     :: "v"(la), "v"(gp) : "memory");
    } else {
        #pragma unroll
        for (int i = tid; i < TILE_N * TILE_K; i += 256) {
            int n = i >> 5, k = i & 31;
            int gk = k0 + k, gn = n0 + n;
            bool ok = (gk < K) && (gn < N);
            size_t gi = BT ? ((size_t)(ok ? gn : 0) * ldb + (ok ? gk : 0))
                           : ((size_t)(ok ? gk : 0) * ldb + (ok ? gn : 0));
            float v = B[gi];
            Bs[n][k] = ok ? v : 0.0f;
        }
    }
}

// ---------------------------------------------------------------------------
// Generic WMMA GEMM:  C = epilogue(A[M,K] @ B + bias + addend)
//   BT = false : B stored [K,N] row-major   -- x @ W
//   BT = true  : B stored [N,K] row-major   -- x @ W^T
// 8 waves / block; each wave owns a 16x16 C tile of a 64x32 macro-tile.
// Double-buffered: tile t+1 is DMA'd (async-to-LDS) while tile t runs WMMA;
// s_wait_asynccnt 0 + barrier at loop top closes the hazard.
// Fragments: ds_load_b128 + v_cvt_pk_bf16_f32 -> v_wmma_f32_16x16x32_bf16.
// ---------------------------------------------------------------------------
template<bool BT>
__global__ __launch_bounds__(256)
void gemm_wmma(const float* __restrict__ A, int lda,
               const float* __restrict__ B, int ldb,
               const float* __restrict__ bias,
               const float* __restrict__ addend, int ld_add,
               float* __restrict__ C, int ldc,
               float* __restrict__ C2,
               int M, int N, int K, int do_relu)
{
    __shared__ __align__(16) float As[2][TILE_M][APAD];
    __shared__ __align__(16) float Bs[2][TILE_N][APAD];

    const int tid  = threadIdx.x;
    const int wave = tid >> 5;
    const int lane = tid & 31;
    const int wm   = (wave & 3) << 4;   // 0,16,32,48
    const int wn   = (wave >> 2) << 4;  // 0,16
    const int m0   = blockIdx.y * TILE_M;
    const int n0   = blockIdx.x * TILE_N;

    v8f acc = {};

    const int nT = (K + TILE_K - 1) / TILE_K;
    stage_tiles<BT>(As[0], Bs[0], A, lda, B, ldb, m0, n0, 0, M, N, K, tid);

    for (int t = 0; t < nT; ++t) {
        const int cur = t & 1;
        asm volatile("s_wait_asynccnt 0x0" ::: "memory");
        __syncthreads();
        if (t + 1 < nT)
            stage_tiles<BT>(As[cur ^ 1], Bs[cur ^ 1], A, lda, B, ldb,
                            m0, n0, (t + 1) * TILE_K, M, N, K, tid);

        // ---- fragments per CDNA5 16-bit WMMA layouts (ISA 7.12.2) ----
        // A: lane (m = lane%16, hi = lane/16) holds K = hi*8 + [0..7] and +16.
        // B: lane (n = lane%16, hi) holds K = hi*8 + [0..7] and +16.
        const int ml = lane & 15;
        const int hi = lane >> 4;
        const int ai = hi << 1;   // (hi*8)/4
        const int bi = hi << 1;   // (hi*8)/4
        const f32x4* arow = (const f32x4*)(&As[cur][wm + ml][0]);
        const f32x4* brow = (const f32x4*)(&Bs[cur][wn + ml][0]);
        f32x4 a0 = arow[ai],     a1 = arow[ai + 1];
        f32x4 a2 = arow[ai + 4], a3 = arow[ai + 5];
        f32x4 b0 = brow[bi],     b1 = brow[bi + 1];
        f32x4 b2 = brow[bi + 4], b3 = brow[bi + 5];
        v16bf af, bv;
        #pragma unroll
        for (int i = 0; i < 4; ++i) {
            af[i]      = (__bf16)a0[i];
            af[4 + i]  = (__bf16)a1[i];
            af[8 + i]  = (__bf16)a2[i];
            af[12 + i] = (__bf16)a3[i];
            bv[i]      = (__bf16)b0[i];
            bv[4 + i]  = (__bf16)b1[i];
            bv[8 + i]  = (__bf16)b2[i];
            bv[12 + i] = (__bf16)b3[i];
        }
        acc = __builtin_amdgcn_wmma_f32_16x16x32_bf16(
                  false, af, false, bv, (short)0, acc, false, false);
        __syncthreads();
    }

    // ---- epilogue: C/D layout — VGPR r -> M = r + 8*(lane>=16), N = lane%16
    const int nl    = lane & 15;
    const int mbase = (lane >> 4) << 3;
    const int gn    = n0 + wn + nl;
    if (gn < N) {
        const float bb = bias ? bias[gn] : 0.0f;
        if (m0 + TILE_M <= M) {          // interior block: unguarded stores
            #pragma unroll
            for (int r = 0; r < 8; ++r) {
                int gm = m0 + wm + mbase + r;
                float v = acc[r] + bb;
                if (addend) v += addend[(size_t)gm * ld_add + gn];
                if (do_relu) v = fmaxf(v, 0.0f);
                C[(size_t)gm * ldc + gn] = v;
                if (C2) C2[(size_t)gm * ldc + gn] = v;
            }
        } else {
            #pragma unroll
            for (int r = 0; r < 8; ++r) {
                int gm = m0 + wm + mbase + r;
                if (gm < M) {
                    float v = acc[r] + bb;
                    if (addend) v += addend[(size_t)gm * ld_add + gn];
                    if (do_relu) v = fmaxf(v, 0.0f);
                    C[(size_t)gm * ldc + gn] = v;
                    if (C2) C2[(size_t)gm * ldc + gn] = v;
                }
            }
        }
    }
}

// ---------------------------------------------------------------------------
// Elementwise kernels: 2D/3D grids (row = blockIdx.x, column chunk =
// blockIdx.y) -> no integer division anywhere, fully coalesced columns.
// ---------------------------------------------------------------------------

// dst[a,:] = (base[a,:]+) sum_q(mb[a2b[a,q],:]) * max_q(mb[a2b[a,q],:])
__global__ void aggregate_kernel(const float* __restrict__ mb,
                                 const int* __restrict__ a2b,
                                 const float* __restrict__ base,
                                 float* __restrict__ dst, int dst_ld,
                                 int H, int nb)
{
    const int a = blockIdx.x;
    const int j = blockIdx.y * blockDim.x + threadIdx.x;
    if (j >= H) return;
    float s = 0.0f, mx = -3.0e38f;
    for (int q = 0; q < nb; ++q) {
        int b = a2b[a * nb + q];
        float v = mb[(size_t)b * H + j];
        s += v;
        mx = fmaxf(mx, v);
    }
    float agg = s * mx;
    if (base) agg += base[(size_t)a * H + j];
    dst[(size_t)a * dst_ld + j] = agg;
}

// tmp[b,:] = ma[b2a[b],:] - mb[b2revb[b],:]
__global__ void bond_update_kernel(const float* __restrict__ ma,
                                   const float* __restrict__ mb,
                                   const int* __restrict__ b2a,
                                   const int* __restrict__ b2revb,
                                   float* __restrict__ tmp, int H)
{
    const int b = blockIdx.x;
    const int j = blockIdx.y * blockDim.x + threadIdx.x;
    if (j >= H) return;
    tmp[(size_t)b * H + j] =
        ma[(size_t)b2a[b] * H + j] - mb[(size_t)b2revb[b] * H + j];
}

__global__ void copy_cols_kernel(const float* __restrict__ src, int src_ld,
                                 float* __restrict__ dst, int dst_ld, int Hc)
{
    const int r = blockIdx.x;
    const int c = blockIdx.y * blockDim.x + threadIdx.x;
    if (c >= Hc) return;
    dst[(size_t)r * dst_ld + c] = src[(size_t)r * src_ld + c];
}

// msg[r,:] = relu(node[r,:] + gru_bias)
__global__ void msg_kernel(const float* __restrict__ node,
                           const float* __restrict__ gbias,
                           float* __restrict__ msg, int H)
{
    const int r = blockIdx.x;
    const int j = blockIdx.y * blockDim.x + threadIdx.x;
    if (j >= H) return;
    msg[(size_t)r * H + j] = fmaxf(node[(size_t)r * H + j] + gbias[j], 0.0f);
}

// h0[b,:] = max_t node[1 + b*T + t, :] ; duplicated to fwd/bwd state buffers
__global__ void h0_kernel(const float* __restrict__ node,
                          float* __restrict__ hf, float* __restrict__ hb,
                          int T, int H)
{
    const int b = blockIdx.x;
    const int j = blockIdx.y * blockDim.x + threadIdx.x;
    if (j >= H) return;
    float mx = -3.0e38f;
    for (int t = 0; t < T; ++t)
        mx = fmaxf(mx, node[(size_t)(1 + b * T + t) * H + j]);
    hf[(size_t)b * H + j] = mx;
    hb[(size_t)b * H + j] = mx;
}

// message600 row 0 = [msg[0,:], msg[0,:]]
__global__ void row0_kernel(const float* __restrict__ msg,
                            float* __restrict__ m600, int H)
{
    int j = blockIdx.x * blockDim.x + threadIdx.x;
    if (j >= 2 * H) return;
    m600[j] = msg[j >= H ? j - H : j];
}

// Fused GRU gates for both directions (blockIdx.z = dir); writes h_new + y.
__global__ void gru_gate_kernel(const float* __restrict__ Xf,
                                const float* __restrict__ Xb,
                                const float* __restrict__ hhf,
                                const float* __restrict__ hhb,
                                const float* __restrict__ hf_in,
                                const float* __restrict__ hb_in,
                                float* __restrict__ hf_out,
                                float* __restrict__ hb_out,
                                float* __restrict__ y,   // rows 1.. of message600
                                int s, int T, int H)
{
    const int b = blockIdx.x;
    const int j = blockIdx.y * blockDim.x + threadIdx.x;
    if (j >= H) return;
    const int dir = blockIdx.z;
    const int t = dir ? (T - 1 - s) : s;

    const float* X    = dir ? Xb : Xf;
    const float* hh   = dir ? hhb : hhf;
    const float* h_in = dir ? hb_in : hf_in;
    float* h_out      = dir ? hb_out : hf_out;

    size_t xr = (size_t)(b * T + t) * (3 * H);
    size_t hr = (size_t)b * (3 * H);
    float ir = X[xr + j], iz = X[xr + H + j], in_ = X[xr + 2 * H + j];
    float gr = hh[hr + j], gz = hh[hr + H + j], gn = hh[hr + 2 * H + j];
    float h  = h_in[(size_t)b * H + j];

    float r = sigmoidf_(ir + gr);
    float z = sigmoidf_(iz + gz);
    float n = tanhf(in_ + r * gn);
    float hnew = (1.0f - z) * n + z * h;

    h_out[(size_t)b * H + j] = hnew;
    y[(size_t)(b * T + t) * (2 * H) + dir * H + j] = hnew;
}

// out[b,:] = mean_t atom_hiddens[1 + b*A + t, :]
__global__ void mean_kernel(const float* __restrict__ ah,
                            float* __restrict__ out, int A, int H)
{
    const int b = blockIdx.x;
    const int j = blockIdx.y * blockDim.x + threadIdx.x;
    if (j >= H) return;
    float s = 0.0f;
    for (int t = 0; t < A; ++t)
        s += ah[(size_t)(1 + b * A + t) * H + j];
    out[(size_t)b * H + j] = s * (1.0f / (float)A);
}

// ---------------------------------------------------------------------------
// Host orchestration
// ---------------------------------------------------------------------------
extern "C" void kernel_launch(void* const* d_in, const int* in_sizes, int n_in,
                              void* d_out, int out_size, void* d_ws, size_t ws_size,
                              hipStream_t stream) {
    (void)n_in; (void)out_size; (void)ws_size;

    const float* f_atoms  = (const float*)d_in[0];
    const float* f_bonds  = (const float*)d_in[1];
    const int*   a2b      = (const int*)d_in[2];
    const int*   b2a      = (const int*)d_in[3];
    const int*   b2revb   = (const int*)d_in[4];
    const float* W_i_atom = (const float*)d_in[5];
    const float* W_i_bond = (const float*)d_in[6];
    const float* W_h      = (const float*)d_in[7];
    const float* W_lr     = (const float*)d_in[8];
    const float* W_o      = (const float*)d_in[9];
    const float* b_o      = (const float*)d_in[10];
    const float* gru_bias = (const float*)d_in[11];
    const float* w_ih_f   = (const float*)d_in[12];
    const float* w_hh_f   = (const float*)d_in[13];
    const float* b_ih_f   = (const float*)d_in[14];
    const float* b_hh_f   = (const float*)d_in[15];
    const float* w_ih_b   = (const float*)d_in[16];
    const float* w_hh_b   = (const float*)d_in[17];
    const float* b_ih_b   = (const float*)d_in[18];
    const float* b_hh_b   = (const float*)d_in[19];

    const int AFD = 133, BFD = 147, H = 300, DEPTH = 5;
    const int Na = in_sizes[0] / AFD;
    const int Nb = in_sizes[1] / BFD;
    const int nb = in_sizes[2] / Na;
    const int A  = 64;
    const int B  = (Na - 1) / A;
    const int T  = A;

    // ---- workspace partition (floats), lifetime-based aliasing ----
    float* ws = (float*)d_ws;
    size_t off = 0;
    auto alloc = [&](size_t n) { size_t o = off; off += n; return o; };
    float* input_atom   = ws + alloc((size_t)Na * H);   // later: atom_hiddens
    float* input_bond   = ws + alloc((size_t)Nb * H);   // later: Xf
    float* message_atom = ws + alloc((size_t)Na * H);   // later: node
    float* message_bond = ws + alloc((size_t)Nb * H);   // later: Xb
    float* r4           = ws + alloc((size_t)Nb * H);   // tmp_bond -> concat900 -> message600
    float* msgbuf       = ws + alloc((size_t)Na * H);
    float* hhf          = ws + alloc((size_t)B * 3 * H);
    float* hhb          = ws + alloc((size_t)B * 3 * H);
    float* hbase        = ws + alloc((size_t)4 * B * H);

    float* tmp_bond     = r4;
    float* concat900    = r4;
    float* message600   = r4;
    float* node         = message_atom;
    float* Xf           = input_bond;
    float* Xb           = message_bond;
    float* atom_hiddens = input_atom;
    float* hf[2] = { hbase,             hbase + (size_t)B * H };
    float* hb[2] = { hbase + 2ull*B*H,  hbase + 3ull*B*H };

    const dim3 blk(256);
    const int HC  = (H + 255) / 256;          // column chunks for H
    const int HC3 = (3 * H + 255) / 256;
    auto gg = [&](int M, int N) {
        return dim3((N + TILE_N - 1) / TILE_N, (M + TILE_M - 1) / TILE_M);
    };

    // 1) input projections (+ duplicate into message buffers)
    gemm_wmma<false><<<gg(Na, H), blk, 0, stream>>>(
        f_atoms, AFD, W_i_atom, H, nullptr, nullptr, 0,
        input_atom, H, message_atom, Na, H, AFD, 1);
    gemm_wmma<false><<<gg(Nb, H), blk, 0, stream>>>(
        f_bonds, BFD, W_i_bond, H, nullptr, nullptr, 0,
        input_bond, H, message_bond, Nb, H, BFD, 1);

    // 2) message passing depths
    for (int d = 0; d < DEPTH - 1; ++d) {
        aggregate_kernel<<<dim3(Na, HC), blk, 0, stream>>>(
            message_bond, a2b, message_atom, message_atom, H, H, nb);
        bond_update_kernel<<<dim3(Nb, HC), blk, 0, stream>>>(
            message_atom, message_bond, b2a, b2revb, tmp_bond, H);
        gemm_wmma<false><<<gg(Nb, H), blk, 0, stream>>>(
            tmp_bond, H, W_h + (size_t)d * H * H, H, nullptr,
            input_bond, H, message_bond, H, nullptr, Nb, H, H, 1);
    }

    // 3) readout: concat [agg, message_atom, input_atom] @ W_lr
    aggregate_kernel<<<dim3(Na, HC), blk, 0, stream>>>(
        message_bond, a2b, nullptr, concat900, 3 * H, H, nb);
    copy_cols_kernel<<<dim3(Na, HC), blk, 0, stream>>>(
        message_atom, H, concat900 + H, 3 * H, H);
    copy_cols_kernel<<<dim3(Na, HC), blk, 0, stream>>>(
        input_atom, H, concat900 + 2 * H, 3 * H, H);
    gemm_wmma<false><<<gg(Na, H), blk, 0, stream>>>(
        concat900, 3 * H, W_lr, H, nullptr, nullptr, 0,
        node, H, nullptr, Na, H, 3 * H, 0);

    // 4) GRU prologue
    msg_kernel<<<dim3(Na, HC), blk, 0, stream>>>(node, gru_bias, msgbuf, H);
    h0_kernel<<<dim3(B, HC), blk, 0, stream>>>(node, hf[0], hb[0], T, H);
    gemm_wmma<true><<<gg(B * T, 3 * H), blk, 0, stream>>>(
        msgbuf + H, H, w_ih_f, H, b_ih_f, nullptr, 0,
        Xf, 3 * H, nullptr, B * T, 3 * H, H, 0);
    gemm_wmma<true><<<gg(B * T, 3 * H), blk, 0, stream>>>(
        msgbuf + H, H, w_ih_b, H, b_ih_b, nullptr, 0,
        Xb, 3 * H, nullptr, B * T, 3 * H, H, 0);
    row0_kernel<<<dim3(HC3), blk, 0, stream>>>(msgbuf, message600, H);

    // 5) bidirectional GRU recurrence (seq dim), y -> message600 rows 1..
    int cur = 0;
    for (int s = 0; s < T; ++s) {
        gemm_wmma<true><<<gg(B, 3 * H), blk, 0, stream>>>(
            hf[cur], H, w_hh_f, H, b_hh_f, nullptr, 0,
            hhf, 3 * H, nullptr, B, 3 * H, H, 0);
        gemm_wmma<true><<<gg(B, 3 * H), blk, 0, stream>>>(
            hb[cur], H, w_hh_b, H, b_hh_b, nullptr, 0,
            hhb, 3 * H, nullptr, B, 3 * H, H, 0);
        gru_gate_kernel<<<dim3(B, HC, 2), blk, 0, stream>>>(
            Xf, Xb, hhf, hhb, hf[cur], hb[cur], hf[1 - cur], hb[1 - cur],
            message600 + 2 * H, s, T, H);
        cur ^= 1;
    }

    // 6) output projection + per-molecule mean
    gemm_wmma<false><<<gg(Na, H), blk, 0, stream>>>(
        message600, 2 * H, W_o, H, b_o, nullptr, 0,
        atom_hiddens, H, nullptr, Na, H, 2 * H, 1);
    mean_kernel<<<dim3(B, HC), blk, 0, stream>>>(
        atom_hiddens, (float*)d_out, A, H);
}
